// SDPAttention_50251117363389
// MI455X (gfx1250) — matrix operand order repaired and checked by reading, not compiled, not run on hardware
//
#include <hip/hip_runtime.h>
#include <stdint.h>

// ---------------------------------------------------------------------------
// Fused attention block for MI455X (gfx1250, wave32, WMMA bf16 16x16x32).
//   B=2, N=2048, C=768, H=12, D=64
// Data path: async global->LDS staging (ASYNCcnt) + WMMA from LDS fragments.
// ---------------------------------------------------------------------------

typedef unsigned short u16;
typedef __attribute__((ext_vector_type(16))) __bf16 v16bf;
typedef __attribute__((ext_vector_type(8)))  float  v8f;
typedef __attribute__((ext_vector_type(4)))  int    v4i;

#define BATCH 2
#define SEQ   2048
#define CDIM  768
#define HEADS 12
#define HD    64

#if __has_builtin(__builtin_amdgcn_global_load_async_to_lds_b128) && \
    __has_builtin(__builtin_amdgcn_s_wait_asynccnt)
#define HAVE_ASYNC 1
#endif

#ifdef HAVE_ASYNC
#define ASYNC_WAIT(n) __builtin_amdgcn_s_wait_asynccnt(n)
#else
#define ASYNC_WAIT(n)
#endif

// 16-byte global -> LDS copy (async engine when available).
// Builtin signature (probed): (v4i AS1*, v4i AS3*, Imm offset, Imm cpol).
__device__ __forceinline__ void copy16(u16* l, const u16* g) {
#ifdef HAVE_ASYNC
  v4i* gg = (v4i*)const_cast<u16*>(g);   // generic reinterpret, drop const
  v4i* ll = (v4i*)l;
  __builtin_amdgcn_global_load_async_to_lds_b128(
      (__attribute__((address_space(1))) v4i*)gg,
      (__attribute__((address_space(3))) v4i*)ll,
      0, 0);
#else
  *(uint4*)l = *(const uint4*)g;
#endif
}

union FragU {
  v16bf    v;
  uint4    u4[2];
  unsigned u[8];
  u16      s[16];
};

__device__ __forceinline__ u16 f2bf(float f) {
  unsigned u = __float_as_uint(f);
  u += 0x7FFFu + ((u >> 16) & 1u);      // round-to-nearest-even
  return (u16)(u >> 16);
}

__device__ __forceinline__ unsigned pk2bf(float lo, float hi) {
  return (unsigned)f2bf(lo) | ((unsigned)f2bf(hi) << 16);
}

// swap lanes l <-> l+16 (wave32): ds_swizzle group-of-32, xor=0x10,and=0x1f
__device__ __forceinline__ float swap16f(float x) {
  return __int_as_float(__builtin_amdgcn_ds_swizzle(__float_as_int(x), 0x401F));
}
__device__ __forceinline__ unsigned swap16u(unsigned x) {
  return (unsigned)__builtin_amdgcn_ds_swizzle((int)x, 0x401F);
}

// A fragment: 16x32 bf16 tile from row-major [rows, ld] source.
// lane<16: row=lane,     K = koff+0..15
// lane>=16: row=lane-16, K = koff+16..31
__device__ __forceinline__ v16bf load_a(const u16* __restrict__ base, int ld,
                                        int koff, int lane) {
  int row = lane & 15;
  int hi  = lane >> 4;
  const u16* p = base + (size_t)row * ld + koff + hi * 16;
  FragU f;
  f.u4[0] = *(const uint4*)(p);
  f.u4[1] = *(const uint4*)(p + 8);
  return f.v;
}

// B fragment: 32x16 bf16 tile where B[k][n] = src[n][k], src row-major.
// lane<16: col=lane, K = koff+0..15 ; lane>=16: col=lane-16, K = koff+16..31
__device__ __forceinline__ v16bf load_b(const u16* __restrict__ base, int ld,
                                        int koff, int lane) {
  int col = lane & 15;
  int hi  = lane >> 4;
  const u16* p = base + (size_t)col * ld + koff + hi * 16;
  FragU f;
  f.u4[0] = *(const uint4*)(p);
  f.u4[1] = *(const uint4*)(p + 8);
  return f.v;
}

#define WMMA_BF16(A, B, C) \
  __builtin_amdgcn_wmma_f32_16x16x32_bf16(false, (A), false, (B), (short)0, (C), false, false)

// ---------------------------------------------------------------------------
__global__ void k_cast(const float* __restrict__ src, u16* __restrict__ dst, int n) {
  int i = blockIdx.x * blockDim.x + threadIdx.x;
  int stride = gridDim.x * blockDim.x;
  for (; i < n; i += stride) dst[i] = f2bf(src[i]);
}

// ---------------------------------------------------------------------------
// QKV GEMM: [4096,768] x [2304,768]^T.  Block = 8 waves sharing one 64-column
// weight strip, staged to LDS (two 384-K halves) via async copies.
// One wave = 16x64 output strip.  Scatter to qb/kb [B*H,N,D] (Q pre-scaled by
// D^-0.5*log2e) and vbT [B*H,D,N].
#define WPITCH 392   // 384 + 8 bf16 pad (bank spread for ds_load_b128)

__global__ void k_qkv(const u16* __restrict__ xb, const u16* __restrict__ wq,
                      u16* __restrict__ qb, u16* __restrict__ kb,
                      u16* __restrict__ vbT) {
  __shared__ u16 sw[64 * WPITCH];            // 50,176 B
  const int tid    = threadIdx.x;
  const int lane   = tid & 31;
  const int wid    = blockIdx.x * (blockDim.x >> 5) + (tid >> 5);
  const int mtile  = wid & 255;              // 256 row tiles
  const int nstrip = wid >> 8;               // 36 strips of 64 (uniform/block)
  const int nbase  = nstrip * 64;
  const u16* arow  = xb + (size_t)mtile * 16 * CDIM;

  // staging slots: thread -> row tid&63, slice (tid>>6)*96, 12 x 16B chunks
  const int crow = tid & 63;
  const int coff = (tid >> 6) * 96;

  v8f acc[4] = {};
  for (int half = 0; half < 2; ++half) {
    const int kb0 = half * 384;
    const u16* wrow = wq + (size_t)(nbase + crow) * CDIM + kb0 + coff;
    u16* lrow = &sw[crow * WPITCH + coff];
#pragma unroll
    for (int i = 0; i < 12; ++i)
      copy16(lrow + i * 8, wrow + i * 8);
    ASYNC_WAIT(0);
    __syncthreads();

#pragma unroll 2
    for (int kk = 0; kk < 384; kk += 32) {
      v16bf a = load_a(arow, CDIM, kb0 + kk, lane);
#pragma unroll
      for (int t = 0; t < 4; ++t) {
        v16bf b = load_b(&sw[(t * 16) * WPITCH], WPITCH, kk, lane);
        acc[t] = WMMA_BF16(a, b, acc[t]);
      }
    }
    __syncthreads();   // before refilling LDS with next half
  }

  const float QS = 0.125f * 1.44269504088896340736f;  // D^-0.5 * log2(e)
  const int hi = lane >> 4;
#pragma unroll
  for (int t = 0; t < 4; ++t) {
    int j     = nbase + t * 16 + (lane & 15);
    int which = j / CDIM;            // 0=Q 1=K 2=V (uniform per wave,t)
    int jm    = j - which * CDIM;
    int h     = jm >> 6;
    int d     = jm & 63;
#pragma unroll
    for (int r = 0; r < 8; ++r) {
      int row = mtile * 16 + r + hi * 8;
      int b_  = row >> 11;           // /2048
      int n_  = row & 2047;
      size_t bh = (size_t)(b_ * HEADS + h);
      float vv = acc[t][r];
      if (which == 0)      qb[(bh * SEQ + n_) * HD + d]  = f2bf(vv * QS);
      else if (which == 1) kb[(bh * SEQ + n_) * HD + d]  = f2bf(vv);
      else                 vbT[(bh * HD + d) * SEQ + n_] = f2bf(vv);
    }
  }
}

// ---------------------------------------------------------------------------
// Flash attention.  Block = 8 waves sharing one (b,h); wave = one 16-query
// tile.  K/V^T 32-key chunks are double-buffered in LDS via async copies.
// S^T = K*Q^T keeps softmax in-lane (one SWAPX16 combine); O^T = V^T * P^T
// reuses the S^T D-layout as the next B-fragment with 8 ds_swizzles.
#define KPITCH 72   // 64 + 8
#define VPITCH 40   // 32 + 8

__global__ void k_flash(const u16* __restrict__ qb, const u16* __restrict__ kb,
                        const u16* __restrict__ vbT, u16* __restrict__ ob) {
  __shared__ u16 sk[2][32 * KPITCH];   // 2 x 4,608 B
  __shared__ u16 sv[2][64 * VPITCH];   // 2 x 5,120 B
  const int tid  = threadIdx.x;
  const int lane = tid & 31;
  const int wid  = blockIdx.x * (blockDim.x >> 5) + (tid >> 5);
  const int qt   = wid & 127;                      // 128 query tiles
  const int bh   = wid >> 7;                       // 0..23 (uniform per block)
  const u16* qp = qb  + (size_t)bh * SEQ * HD + (size_t)qt * 16 * HD;
  const u16* kp = kb  + (size_t)bh * SEQ * HD;
  const u16* vp = vbT + (size_t)bh * HD * SEQ;

  const v16bf bq0 = load_b(qp, HD, 0, lane);       // Q^T, d 0..31
  const v16bf bq1 = load_b(qp, HD, 32, lane);      // Q^T, d 32..63

  // per-thread copy slots: one K 16B chunk + one V^T 16B chunk per buffer
  const int krow = tid >> 3, koff = (tid & 7) * 8;
  const int vrow = tid >> 2, voff = (tid & 3) * 8;

  // prologue: stage chunk 0
  copy16(&sk[0][krow * KPITCH + koff], kp + (size_t)krow * HD + koff);
  copy16(&sv[0][vrow * VPITCH + voff], vp + (size_t)vrow * SEQ + voff);

  v8f o[4] = {};
  float mrun = -__builtin_inff();
  float lrun = 0.f;
  const int hi = lane >> 4;

  for (int kc = 0; kc < SEQ; kc += 32) {
    const int cur = (kc >> 5) & 1;
    if (kc + 32 < SEQ) {               // stage next chunk into other buffer
      const int nb = cur ^ 1;
      const int kn = kc + 32;
      copy16(&sk[nb][krow * KPITCH + koff], kp + (size_t)(kn + krow) * HD + koff);
      copy16(&sv[nb][vrow * VPITCH + voff], vp + (size_t)vrow * SEQ + kn + voff);
      ASYNC_WAIT(2);                   // current chunk's 2 copies complete
    } else {
      ASYNC_WAIT(0);
    }
    __syncthreads();

    const u16* skb = sk[cur];
    const u16* svb = sv[cur];
    v16bf a00 = load_a(skb, KPITCH, 0, lane);
    v16bf a01 = load_a(skb, KPITCH, 32, lane);
    v16bf a10 = load_a(skb + 16 * KPITCH, KPITCH, 0, lane);
    v16bf a11 = load_a(skb + 16 * KPITCH, KPITCH, 32, lane);

    v8f s0 = {};  // keys kc..kc+15 (transposed: col=query)
    v8f s1 = {};  // keys kc+16..kc+31
    s0 = WMMA_BF16(a00, bq0, s0);
    s0 = WMMA_BF16(a01, bq1, s0);
    s1 = WMMA_BF16(a10, bq0, s1);
    s1 = WMMA_BF16(a11, bq1, s1);

    // online softmax (log2 domain; scale folded into Q)
    float mloc = s0[0];
#pragma unroll
    for (int r = 1; r < 8; ++r) mloc = fmaxf(mloc, s0[r]);
#pragma unroll
    for (int r = 0; r < 8; ++r) mloc = fmaxf(mloc, s1[r]);
    mloc = fmaxf(mloc, swap16f(mloc));
    float mnew  = fmaxf(mrun, mloc);
    float alpha = __builtin_amdgcn_exp2f(mrun - mnew);

    float p0[8], p1[8], lsum = 0.f;
#pragma unroll
    for (int r = 0; r < 8; ++r) { p0[r] = __builtin_amdgcn_exp2f(s0[r] - mnew); lsum += p0[r]; }
#pragma unroll
    for (int r = 0; r < 8; ++r) { p1[r] = __builtin_amdgcn_exp2f(s1[r] - mnew); lsum += p1[r]; }
    lsum += swap16f(lsum);
    lrun = lrun * alpha + lsum;
    mrun = mnew;
#pragma unroll
    for (int t = 0; t < 4; ++t) o[t] = o[t] * alpha;

    // Build P^T B-fragment (32 keys x 16 queries) from the S^T D-layout.
    unsigned q0[4], q1[4];
#pragma unroll
    for (int r = 0; r < 4; ++r) {
      q0[r] = pk2bf(p0[2 * r], p0[2 * r + 1]);
      q1[r] = pk2bf(p1[2 * r], p1[2 * r + 1]);
    }
    FragU pb;
#pragma unroll
    for (int r = 0; r < 4; ++r) {
      unsigned sw0 = swap16u(q0[r]);   // partner's keys for this query column
      unsigned sw1 = swap16u(q1[r]);
      pb.u[r]     = hi ? sw1 : q0[r];  // K 0..7   | K 16..23
      pb.u[r + 4] = hi ? q1[r] : sw0;  // K 8..15  | K 24..31
    }

#pragma unroll
    for (int t = 0; t < 4; ++t) {      // O^T += V^T * P^T
      v16bf av = load_a(svb + t * 16 * VPITCH, VPITCH, 0, lane);
      o[t] = WMMA_BF16(av, pb.v, o[t]);
    }
    __syncthreads();                   // done reading before buffer reuse
  }

  const float inv = __builtin_amdgcn_rcpf(lrun);
  const int b_ = bh / HEADS;
  const int h  = bh % HEADS;
  const int q  = qt * 16 + (lane & 15);
  const size_t rowbase = ((size_t)(b_ * SEQ + q)) * CDIM + h * HD;
#pragma unroll
  for (int t = 0; t < 4; ++t)
#pragma unroll
    for (int r = 0; r < 8; ++r) {
      int d = t * 16 + r + hi * 8;
      ob[rowbase + d] = f2bf(o[t][r] * inv);
    }
}

// ---------------------------------------------------------------------------
// Projection GEMM: [4096,768] x [768,768]^T + bias, f32 out.  Same LDS-staged
// weight-strip structure as k_qkv.
__global__ void k_proj(const u16* __restrict__ ob, const u16* __restrict__ wp,
                       const float* __restrict__ bias, float* __restrict__ out) {
  __shared__ u16 sw[64 * WPITCH];
  const int tid    = threadIdx.x;
  const int lane   = tid & 31;
  const int wid    = blockIdx.x * (blockDim.x >> 5) + (tid >> 5);
  const int mtile  = wid & 255;    // 256 row tiles
  const int nstrip = wid >> 8;     // 12 strips of 64 (uniform per block)
  const int nbase  = nstrip * 64;
  const u16* arow  = ob + (size_t)mtile * 16 * CDIM;

  const int crow = tid & 63;
  const int coff = (tid >> 6) * 96;

  v8f acc[4] = {};
  for (int half = 0; half < 2; ++half) {
    const int kb0 = half * 384;
    const u16* wrow = wp + (size_t)(nbase + crow) * CDIM + kb0 + coff;
    u16* lrow = &sw[crow * WPITCH + coff];
#pragma unroll
    for (int i = 0; i < 12; ++i)
      copy16(lrow + i * 8, wrow + i * 8);
    ASYNC_WAIT(0);
    __syncthreads();

#pragma unroll 2
    for (int kk = 0; kk < 384; kk += 32) {
      v16bf a = load_a(arow, CDIM, kb0 + kk, lane);
#pragma unroll
      for (int t = 0; t < 4; ++t) {
        v16bf b = load_b(&sw[(t * 16) * WPITCH], WPITCH, kk, lane);
        acc[t] = WMMA_BF16(a, b, acc[t]);
      }
    }
    __syncthreads();
  }

  const int hi = lane >> 4;
#pragma unroll
  for (int t = 0; t < 4; ++t) {
    int col  = nbase + t * 16 + (lane & 15);
    float bv = bias[col];
#pragma unroll
    for (int r = 0; r < 8; ++r) {
      int row = mtile * 16 + r + hi * 8;
      out[(size_t)row * CDIM + col] = acc[t][r] + bv;
    }
  }
}

// ---------------------------------------------------------------------------
extern "C" void kernel_launch(void* const* d_in, const int* in_sizes, int n_in,
                              void* d_out, int out_size, void* d_ws, size_t ws_size,
                              hipStream_t stream) {
  const float* x      = (const float*)d_in[0];   // [2,2048,768]
  const float* w_qkv  = (const float*)d_in[1];   // [2304,768]
  const float* w_proj = (const float*)d_in[2];   // [768,768]
  const float* b_proj = (const float*)d_in[3];   // [768]
  float* out = (float*)d_out;                    // [2,2048,768]

  const size_t XN = (size_t)BATCH * SEQ * CDIM;        // 3,145,728
  const size_t WQ = (size_t)3 * CDIM * CDIM;           // 1,769,472
  const size_t WP = (size_t)CDIM * CDIM;               //   589,824
  const size_t QK = (size_t)BATCH * HEADS * SEQ * HD;  // 3,145,728

  u16* ws     = (u16*)d_ws;
  u16* xb     = ws;              // bf16 activations
  u16* wqkvb  = xb + XN;
  u16* wprojb = wqkvb + WQ;
  u16* qb     = wprojb + WP;     // [B*H, N, D], pre-scaled
  u16* kb     = qb + QK;         // [B*H, N, D]
  u16* vbT    = kb + QK;         // [B*H, D, N] (transposed)
  u16* obuf   = vbT + QK;        // attention out, [B*N, C] bf16

  k_cast<<<2048, 256, 0, stream>>>(x, xb, (int)XN);
  k_cast<<<2048, 256, 0, stream>>>(w_qkv, wqkvb, (int)WQ);
  k_cast<<<512,  256, 0, stream>>>(w_proj, wprojb, (int)WP);

  // 256 mtiles * 36 nstrips = 9216 wave tasks / 8 waves per block
  k_qkv<<<1152, 256, 0, stream>>>(xb, wqkvb, qb, kb, vbT);

  // 2*12*128 = 3072 wave tasks
  k_flash<<<384, 256, 0, stream>>>(qb, kb, vbT, obuf);

  // 256 mtiles * 12 nstrips = 3072 wave tasks
  k_proj<<<384, 256, 0, stream>>>(obuf, wprojb, b_proj, out);
}
